// Seq2SeqAttentionModel_23648089932156
// MI455X (gfx1250) — matrix operand order repaired and checked by reading, not compiled, not run on hardware
//
#include <hip/hip_runtime.h>
#include <hip/hip_bf16.h>
#include <cstdint>

typedef __bf16 bf16;
typedef __attribute__((ext_vector_type(16))) __bf16 v16bf;
typedef __attribute__((ext_vector_type(8)))  float  v8f;

#define PAD_INDEX 0
#define START_INDEX 1

__device__ __forceinline__ bf16 to_bf16(float f) { return (bf16)f; }
__device__ __forceinline__ float sigm(float x) { return 1.0f / (1.0f + expf(-x)); }

// ---------------------------------------------------------------------------
// f32 -> bf16 elementwise convert
// ---------------------------------------------------------------------------
__global__ void k_f32_to_bf16(const float* __restrict__ in, bf16* __restrict__ out, int n) {
    int i = blockIdx.x * blockDim.x + threadIdx.x;
    if (i < n) out[i] = to_bf16(in[i]);
}

// ---------------------------------------------------------------------------
// Embedding gathers (already transposed to [step, batch, E] row-major, bf16)
// ---------------------------------------------------------------------------
__global__ void k_embed_enc(const int* __restrict__ tok, const float* __restrict__ emb,
                            bf16* __restrict__ x, int B, int S, int E) {
    int i = blockIdx.x * blockDim.x + threadIdx.x;
    int total = B * S * E;
    if (i >= total) return;
    int e = i % E; int row = i / E;
    int s = row / B; int b = row % B;
    int t = tok[b * S + s];
    x[i] = to_bf16(emb[(size_t)t * E + e]);
}

__global__ void k_embed_dec(const int* __restrict__ tgt, const float* __restrict__ emb,
                            bf16* __restrict__ x, int B, int T, int E) {
    int i = blockIdx.x * blockDim.x + threadIdx.x;
    int total = B * T * E;
    if (i >= total) return;
    int e = i % E; int row = i / E;
    int ts = row / B; int b = row % B;
    int t = (ts == 0) ? START_INDEX : tgt[b * T + (ts - 1)];
    x[i] = to_bf16(emb[(size_t)t * E + e]);
}

// ---------------------------------------------------------------------------
// Zero hidden state (f32 + bf16 copies)
// ---------------------------------------------------------------------------
__global__ void k_zero_h(float* __restrict__ h, bf16* __restrict__ hb, int n) {
    int i = blockIdx.x * blockDim.x + threadIdx.x;
    if (i < n) { h[i] = 0.0f; hb[i] = to_bf16(0.0f); }
}

// ---------------------------------------------------------------------------
// WMMA GEMM, register-blocked 4x in M:
//   out[M,N] = A[M,K](bf16,row-major) @ W[N,K]^T(bf16,row-major) + bias[N]
// One wave owns a 64x16 output strip (4 M-tiles x 1 N-tile). Per 32-wide
// k-step the wave loads ONE B fragment (2x b128) and FOUR A fragments
// (4 x 2x b128), issuing 4 v_wmma_f32_16x16x32_bf16 against the shared B.
// B traffic is cut 4x vs one-tile-per-wave; 2.5 b128 loads per WMMA.
// Requires M % 64 == 0, N % 16 == 0, K % 32 == 0 (true for every GEMM here).
// Optional outb: bf16 copy of the result (fuses the downstream convert).
// ---------------------------------------------------------------------------
__global__ void __launch_bounds__(256)
k_wmma_gemm4(const bf16* __restrict__ A, int lda,
             const bf16* __restrict__ W, int ldw,
             const float* __restrict__ bias,
             float* __restrict__ out, long long ldout,
             bf16* __restrict__ outb,
             int M, int N, int K) {
    int wid = blockIdx.x * (blockDim.x >> 5) + (threadIdx.x >> 5);
    int nms = M >> 6;                         // 64-row strips
    int nnt = N >> 4;
    if (wid >= nms * nnt) return;             // wave-uniform: EXEC stays all-1s
    int ms = wid % nms, nt = wid / nms;
    int lane = threadIdx.x & 31;
    int half = lane >> 4;
    int l = lane & 15;

    const bf16* a0 = A + (size_t)((ms << 6) + l) * lda + (half << 3);
    const bf16* brow = W + (size_t)((nt << 4) + l) * ldw + (half << 3);
    size_t astride = (size_t)16 * lda;        // next M-tile

    v8f acc0 = {}, acc1 = {}, acc2 = {}, acc3 = {};
    for (int kb = 0; kb < K; kb += 32) {
        union { uint4 u[2]; v16bf v; } fb, f0, f1, f2, f3;
        const uint4* pb = reinterpret_cast<const uint4*>(brow + kb);
        fb.u[0] = pb[0];                      // K = kb+h*8 .. +7
        fb.u[1] = pb[2];                      // K = kb+16+h*8 .. +7
        const uint4* p0 = reinterpret_cast<const uint4*>(a0 + kb);
        const uint4* p1 = reinterpret_cast<const uint4*>(a0 + astride + kb);
        const uint4* p2 = reinterpret_cast<const uint4*>(a0 + 2 * astride + kb);
        const uint4* p3 = reinterpret_cast<const uint4*>(a0 + 3 * astride + kb);
        f0.u[0] = p0[0]; f0.u[1] = p0[2];     // K = kb+h*8 .. , kb+16+h*8 ..
        f1.u[0] = p1[0]; f1.u[1] = p1[2];
        f2.u[0] = p2[0]; f2.u[1] = p2[2];
        f3.u[0] = p3[0]; f3.u[1] = p3[2];
        acc0 = __builtin_amdgcn_wmma_f32_16x16x32_bf16(false, f0.v, false, fb.v, (short)0, acc0, false, false);
        acc1 = __builtin_amdgcn_wmma_f32_16x16x32_bf16(false, f1.v, false, fb.v, (short)0, acc1, false, false);
        acc2 = __builtin_amdgcn_wmma_f32_16x16x32_bf16(false, f2.v, false, fb.v, (short)0, acc2, false, false);
        acc3 = __builtin_amdgcn_wmma_f32_16x16x32_bf16(false, f3.v, false, fb.v, (short)0, acc3, false, false);
    }

    int colg = (nt << 4) + l;
    float bv = bias ? bias[colg] : 0.0f;
    int rbase = (ms << 6) + (half << 3);      // rows for acc element j: rbase + i*16 + j
#pragma unroll
    for (int i = 0; i < 4; ++i) {
        v8f acc = (i == 0) ? acc0 : (i == 1) ? acc1 : (i == 2) ? acc2 : acc3;
#pragma unroll
        for (int j = 0; j < 8; ++j) {
            float v = acc[j] + bv;
            size_t o = (size_t)(rbase + i * 16 + j) * ldout + colg;
            out[o] = v;
            if (outb) outb[o] = to_bf16(v);
        }
    }
}

// ---------------------------------------------------------------------------
// GRU gate combine:  gi includes bih, gh includes bhh (added in their GEMMs).
//   r = sig(gi_r+gh_r); z = sig(gi_z+gh_z); n = tanh(gi_n + r*gh_n)
//   h2 = (1-z)*n + z*hprev
// Writes h2 as f32 + bf16; optionally also into enc_h[(b*S+s)*H+i] (f32+bf16).
// ---------------------------------------------------------------------------
__global__ void k_gru_gate(const float* __restrict__ gi, const float* __restrict__ gh,
                           const float* __restrict__ hprev,
                           float* __restrict__ hout, bf16* __restrict__ houtb,
                           float* __restrict__ ench, bf16* __restrict__ enchb,
                           int sstep, int S, int B, int H) {
    int idx = blockIdx.x * blockDim.x + threadIdx.x;
    if (idx >= B * H) return;
    int b = idx / H, i = idx % H;
    const float* gr = gi + (size_t)b * 3 * H;
    const float* hr = gh + (size_t)b * 3 * H;
    float r = sigm(gr[i] + hr[i]);
    float z = sigm(gr[H + i] + hr[H + i]);
    float n = tanhf(gr[2 * H + i] + r * hr[2 * H + i]);
    float h2 = (1.0f - z) * n + z * hprev[idx];
    hout[idx] = h2;
    houtb[idx] = to_bf16(h2);
    if (ench) {
        size_t o = ((size_t)b * S + sstep) * H + i;
        ench[o] = h2;
        enchb[o] = to_bf16(h2);
    }
}

// ---------------------------------------------------------------------------
// Attention (one block per batch row b, 256 threads):
//   scores[s] = sum_h att_v[h]*tanh(att_pre[b,s,h] + ad[b,h])
//   e = exp(scores)*mask; w = e/(sum+eps); ctx[h] = sum_s w[s]*enc_h[b,s,h]
// Also packs comb input: combA[b] = concat(h_bf16[b], bf16(ctx))
// ---------------------------------------------------------------------------
__global__ void k_attention(const float* __restrict__ attpre, const float* __restrict__ ad,
                            const float* __restrict__ attv, const float* __restrict__ ench,
                            const int* __restrict__ inputs, const bf16* __restrict__ hb,
                            bf16* __restrict__ combA, int S, int H, int B) {
    int b = blockIdx.x;
    int tid = threadIdx.x;
    __shared__ float part[256];
    __shared__ float wsm[64];
    __shared__ float sinv;

    // pack previous hidden into first half of comb input
    for (int i = tid; i < H; i += 256)
        combA[(size_t)b * 2 * H + i] = hb[(size_t)b * H + i];

    // scores: thread (s,q) reduces 128 h-values
    int s = tid >> 2, q = tid & 3;
    const float* prow = attpre + ((size_t)b * S + s) * H;
    const float* adr = ad + (size_t)b * H;
    float acc = 0.0f;
    int h0 = q * 128;
    for (int h = h0; h < h0 + 128; ++h)
        acc += attv[h] * tanhf(prow[h] + adr[h]);
    part[tid] = acc;
    __syncthreads();

    if (tid < 64) {
        float sc = part[4 * tid] + part[4 * tid + 1] + part[4 * tid + 2] + part[4 * tid + 3];
        wsm[tid] = (inputs[b * S + tid] != PAD_INDEX) ? expf(sc) : 0.0f;
    }
    __syncthreads();
    if (tid == 0) {
        float sum = 0.0f;
        for (int i = 0; i < S; ++i) sum += wsm[i];
        sinv = 1.0f / (sum + 1e-6f);
    }
    __syncthreads();

    // ctx into second half of comb input
    for (int h = tid; h < H; h += 256) {
        float c = 0.0f;
        for (int s2 = 0; s2 < S; ++s2)
            c += wsm[s2] * ench[((size_t)b * S + s2) * H + h];
        combA[(size_t)b * 2 * H + H + h] = to_bf16(c * sinv);
    }
}

// ---------------------------------------------------------------------------
// Host-side orchestration
// ---------------------------------------------------------------------------
extern "C" void kernel_launch(void* const* d_in, const int* in_sizes, int n_in,
                              void* d_out, int out_size, void* d_ws, size_t ws_size,
                              hipStream_t stream) {
    const int V = 32000, E = 256, H = 512, B = 64, S = 64, T = 32;
    const int G = 3 * H;  // 1536

    const int*   inputs   = (const int*)d_in[0];
    const int*   targets  = (const int*)d_in[1];
    const float* emb      = (const float*)d_in[2];
    const float* enc_Wih  = (const float*)d_in[3];
    const float* enc_Whh  = (const float*)d_in[4];
    const float* enc_bih  = (const float*)d_in[5];
    const float* enc_bhh  = (const float*)d_in[6];
    const float* dec_Wih  = (const float*)d_in[7];
    const float* dec_Whh  = (const float*)d_in[8];
    const float* dec_bih  = (const float*)d_in[9];
    const float* dec_bhh  = (const float*)d_in[10];
    const float* attW_enc = (const float*)d_in[11];
    const float* attb_enc = (const float*)d_in[12];
    const float* attW_dec = (const float*)d_in[13];
    const float* attb_dec = (const float*)d_in[14];
    const float* att_red  = (const float*)d_in[15];
    const float* combW    = (const float*)d_in[16];
    const float* combb    = (const float*)d_in[17];
    const float* projW    = (const float*)d_in[18];
    const float* projb    = (const float*)d_in[19];
    float* outp = (float*)d_out;

    char* ws = (char*)d_ws;
    size_t off = 0;
    auto alloc = [&](size_t bytes) -> void* {
        void* p = ws + off;
        off = (off + bytes + 255) & ~(size_t)255;
        return p;
    };

    bf16* b_encWih  = (bf16*)alloc((size_t)G * E * 2);
    bf16* b_encWhh  = (bf16*)alloc((size_t)G * H * 2);
    bf16* b_decWih  = (bf16*)alloc((size_t)G * E * 2);
    bf16* b_decWhh  = (bf16*)alloc((size_t)G * H * 2);
    bf16* b_attWenc = (bf16*)alloc((size_t)H * H * 2);
    bf16* b_attWdec = (bf16*)alloc((size_t)H * H * 2);
    bf16* b_combW   = (bf16*)alloc((size_t)H * 2 * H * 2);
    bf16* b_projW   = (bf16*)alloc((size_t)V * H * 2);
    bf16* x_enc     = (bf16*)alloc((size_t)S * B * E * 2);
    bf16* x_dec     = (bf16*)alloc((size_t)T * B * E * 2);
    float* gi_enc   = (float*)alloc((size_t)S * B * G * 4);
    float* gi_dec   = (float*)alloc((size_t)T * B * G * 4);
    float* gh       = (float*)alloc((size_t)B * G * 4);
    float* h_f      = (float*)alloc((size_t)B * H * 4);
    bf16*  h_bf     = (bf16*)alloc((size_t)B * H * 2);
    float* enc_h    = (float*)alloc((size_t)B * S * H * 4);
    bf16*  enc_hb   = (bf16*)alloc((size_t)B * S * H * 2);
    float* att_pre  = (float*)alloc((size_t)B * S * H * 4);
    float* ad       = (float*)alloc((size_t)B * H * 4);
    bf16*  combA    = (bf16*)alloc((size_t)B * 2 * H * 2);
    float* comb_f   = (float*)alloc((size_t)B * H * 4);
    bf16*  comb_bf  = (bf16*)alloc((size_t)B * H * 2);
    (void)ws_size; (void)in_sizes; (void)n_in; (void)out_size;

    auto cvt = [&](const float* src, bf16* dst, int n) {
        k_f32_to_bf16<<<(n + 255) / 256, 256, 0, stream>>>(src, dst, n);
    };
    // out[M,N] = A @ W^T + bias ; optional fused bf16 copy of out
    auto gemm = [&](const bf16* A, int lda, const bf16* W, int ldw, const float* bias,
                    float* out, long long ldout, bf16* outb, int M, int N, int K) {
        int waves = (M >> 6) * (N >> 4);
        int blocks = (waves + 7) / 8;               // 8 waves (256 threads) per block
        k_wmma_gemm4<<<blocks, 256, 0, stream>>>(A, lda, W, ldw, bias, out, ldout, outb,
                                                 M, N, K);
    };

    // ---- one-time weight conversion to bf16 ----
    cvt(enc_Wih, b_encWih, G * E);
    cvt(enc_Whh, b_encWhh, G * H);
    cvt(dec_Wih, b_decWih, G * E);
    cvt(dec_Whh, b_decWhh, G * H);
    cvt(attW_enc, b_attWenc, H * H);
    cvt(attW_dec, b_attWdec, H * H);
    cvt(combW, b_combW, H * 2 * H);
    cvt(projW, b_projW, V * H);

    // ---- embedding gathers ----
    k_embed_enc<<<(B * S * E + 255) / 256, 256, 0, stream>>>(inputs, emb, x_enc, B, S, E);
    k_embed_dec<<<(B * T * E + 255) / 256, 256, 0, stream>>>(targets, emb, x_dec, B, T, E);

    // ---- hoisted batched GEMMs: gi for all steps (x-side of both GRUs) ----
    gemm(x_enc, E, b_encWih, E, enc_bih, gi_enc, G, nullptr, S * B, G, E);
    gemm(x_dec, E, b_decWih, E, dec_bih, gi_dec, G, nullptr, T * B, G, E);

    // ---- encoder scan ----
    k_zero_h<<<(B * H + 255) / 256, 256, 0, stream>>>(h_f, h_bf, B * H);
    for (int s = 0; s < S; ++s) {
        gemm(h_bf, H, b_encWhh, H, enc_bhh, gh, G, nullptr, B, G, H);
        k_gru_gate<<<(B * H + 255) / 256, 256, 0, stream>>>(
            gi_enc + (size_t)s * B * G, gh, h_f, h_f, h_bf, enc_h, enc_hb, s, S, B, H);
    }

    // ---- hoisted attention-encoder projection (step-invariant) ----
    gemm(enc_hb, H, b_attWenc, H, attb_enc, att_pre, H, nullptr, B * S, H, H);

    // ---- decoder scan ----
    k_zero_h<<<(B * H + 255) / 256, 256, 0, stream>>>(h_f, h_bf, B * H);
    for (int t = 0; t < T; ++t) {
        gemm(h_bf, H, b_attWdec, H, attb_dec, ad, H, nullptr, B, H, H);
        k_attention<<<B, 256, 0, stream>>>(att_pre, ad, att_red, enc_h, inputs,
                                           h_bf, combA, S, H, B);
        // comb GEMM with fused f32->bf16 epilogue (comb_bf feeds the next GEMM)
        gemm(combA, 2 * H, b_combW, 2 * H, combb, comb_f, H, comb_bf, B, H, 2 * H);
        gemm(comb_bf, H, b_decWhh, H, dec_bhh, gh, G, nullptr, B, G, H);
        k_gru_gate<<<(B * H + 255) / 256, 256, 0, stream>>>(
            gi_dec + (size_t)t * B * G, gh, comb_f, h_f, h_bf,
            nullptr, nullptr, 0, S, B, H);
        // dominant GEMM: vocabulary projection, written straight into d_out[b][t][:]
        gemm(h_bf, H, b_projW, H, projb, outp + (size_t)t * V, (long long)T * V, nullptr,
             B, V, H);
    }
}
